// EdgeUpdate_50173807952925
// MI455X (gfx1250) — matrix operand order repaired and checked by reading, not compiled, not run on hardware
//
#include <hip/hip_runtime.h>
#include <hip/hip_bf16.h>

#define CC 16
#define DD 512           // 2*C*C
#define NC2 1088         // 4*C*C + 4*C

typedef __attribute__((ext_vector_type(16))) _Float16 v16h;
typedef __attribute__((ext_vector_type(8)))  _Float16 v8h;
typedef __attribute__((ext_vector_type(8)))  float    v8f;

__device__ __forceinline__ v8f wmma_f16(v16h a, v16h b, v8f c) {
  return __builtin_amdgcn_wmma_f32_16x16x32_f16(
      false, a, false, b, (short)0, c, false, false);
}

// ---------------------------------------------------------------------------
// Kernel 0: convert + transpose W1 (512x512) and W2 (512x1088) to f16,
// stored column-major (W^T) so each lane's B-fragment is 16 contiguous halves.
// ---------------------------------------------------------------------------
__global__ void prep_weights(const float* __restrict__ W1,
                             const float* __restrict__ W2,
                             _Float16* __restrict__ W1T,
                             _Float16* __restrict__ W2T) {
  const int total1 = DD * DD;
  const int total2 = NC2 * DD;
  int i = blockIdx.x * blockDim.x + threadIdx.x;
  if (i < total1) {
    int n = i / DD, k = i % DD;
    W1T[(size_t)n * DD + k] = (_Float16)W1[(size_t)k * DD + n];
  } else if (i < total1 + total2) {
    int t = i - total1;
    int n = t / DD, k = t % DD;
    W2T[(size_t)n * DD + k] = (_Float16)W2[(size_t)k * NC2 + n];
  }
}

// ---------------------------------------------------------------------------
// Kernel 1: per-edge overlaps  h = [A^T M B ; xa^T M xb]  (f16, E x 512)
// One wave per edge; all five 16x16 matrices staged TRANSPOSED in LDS so
// every WMMA fragment build is contiguous ds_load_b128 + v_cvt_pk_f16_f32.
// ---------------------------------------------------------------------------
__global__ __launch_bounds__(256)
void overlap_kernel(const float* __restrict__ x, const int* __restrict__ eidx,
                    const float* __restrict__ efa, const float* __restrict__ efb,
                    const float* __restrict__ em, _Float16* __restrict__ hbuf,
                    int E_) {
  __shared__ __align__(16) float lT[8][5][256];  // 0:M^T 1:A^T 2:B^T 3:Xa^T 4:Xb^T
  __shared__ __align__(16) float lP[8][256];     // intermediate P, row-major

  const int tid = threadIdx.x, lane = tid & 31, wave = tid >> 5;
  const int e = blockIdx.x * 8 + wave;
  const bool valid = (e < E_);
  const int half = lane >> 4;
  const int lr = lane & 15;
  const int kb = half * 8;

  int ia = 0, ib = 0;
  if (valid) { ia = eidx[e]; ib = eidx[E_ + e]; }

  if (valid) {
    const float4* srcs[5] = {
      (const float4*)(em  + (size_t)e  * 256),
      (const float4*)(efa + (size_t)e  * 256),
      (const float4*)(efb + (size_t)e  * 256),
      (const float4*)(x   + (size_t)ia * 256),
      (const float4*)(x   + (size_t)ib * 256)
    };
    #pragma unroll
    for (int m = 0; m < 5; ++m) {
      float* dst = lT[wave][m];
      #pragma unroll
      for (int t = 0; t < 2; ++t) {
        const int f4 = lane + 32 * t;
        const float4 v = srcs[m][f4];
        const int b0 = f4 * 4;
        dst[((b0 + 0) & 15) * 16 + ((b0 + 0) >> 4)] = v.x;
        dst[((b0 + 1) & 15) * 16 + ((b0 + 1) >> 4)] = v.y;
        dst[((b0 + 2) & 15) * 16 + ((b0 + 2) >> 4)] = v.z;
        dst[((b0 + 3) & 15) * 16 + ((b0 + 3) >> 4)] = v.w;
      }
    }
  }
  __syncthreads();

  // A-fragment from a row-contiguous f32 16x16: elements 0..7 = m[lr*16+kb..+7]
  auto fragA = [&](const float* m) -> v16h {
    const float4* p = (const float4*)(m + lr * 16 + kb);
    const float4 u0 = p[0], u1 = p[1];
    v16h f = {};
    f[0] = (_Float16)u0.x; f[1] = (_Float16)u0.y;
    f[2] = (_Float16)u0.z; f[3] = (_Float16)u0.w;
    f[4] = (_Float16)u1.x; f[5] = (_Float16)u1.y;
    f[6] = (_Float16)u1.z; f[7] = (_Float16)u1.w;
    return f;
  };
  // B-fragment from transposed f32 (mT[col*16+k]): elements 0..7 = mT[lr*16+kb..+7].
  auto fragB = [&](const float* mT) -> v16h {
    const float4* p = (const float4*)(mT + lr * 16 + kb);
    const float4 u0 = p[0], u1 = p[1];
    v16h f = {};
    f[0]  = (_Float16)u0.x; f[1]  = (_Float16)u0.y;
    f[2]  = (_Float16)u0.z; f[3]  = (_Float16)u0.w;
    f[4]  = (_Float16)u1.x; f[5]  = (_Float16)u1.y;
    f[6]  = (_Float16)u1.z; f[7]  = (_Float16)u1.w;
    return f;   // K rows 16..31 are zero padding
  };

  const float* MT  = lT[wave][0];
  const float* AT  = lT[wave][1];
  const float* BT  = lT[wave][2];
  const float* XaT = lT[wave][3];
  const float* XbT = lT[wave][4];
  const v8f z = {};
  _Float16* hr = hbuf + (size_t)e * DD;

  // ---- edge overlap: O = (A^T M) B ----
  // P = A^T @ M : A_op[c][i] = A[i][c] = AT[c*16+i] (row-contiguous -> fragA)
  //               B_op[i][j] = M[i][j] = MT[j*16+i] (fragB on MT)
  v8f p = wmma_f16(fragA(AT), fragB(MT), z);
  #pragma unroll
  for (int v = 0; v < 8; ++v) lP[wave][(v + half * 8) * 16 + lr] = p[v];
  __syncthreads();
  // O = P @ B : A_op[c][j] = P[c][j] row-major; B_op[j][m] = B[j][m] = BT[m*16+j]
  v8f o = wmma_f16(fragA(lP[wave]), fragB(BT), z);
  if (valid) {
    #pragma unroll
    for (int v = 0; v < 8; ++v) hr[(v + half * 8) * 16 + lr] = (_Float16)o[v];
  }
  __syncthreads();

  // ---- node overlap: O = (xa^T M) xb ----
  p = wmma_f16(fragA(XaT), fragB(MT), z);
  #pragma unroll
  for (int v = 0; v < 8; ++v) lP[wave][(v + half * 8) * 16 + lr] = p[v];
  __syncthreads();
  o = wmma_f16(fragA(lP[wave]), fragB(XbT), z);
  if (valid) {
    #pragma unroll
    for (int v = 0; v < 8; ++v) hr[256 + (v + half * 8) * 16 + lr] = (_Float16)o[v];
  }
}

// ---------------------------------------------------------------------------
// Kernel 2: h2 = layernorm(silu(h @ W1 + b1)) * g + b, in place (f16).
// ---------------------------------------------------------------------------
__global__ __launch_bounds__(256)
void gemm1_ln_kernel(_Float16* __restrict__ hbuf,
                     const _Float16* __restrict__ W1T,
                     const float* __restrict__ b1,
                     const float* __restrict__ ln_g,
                     const float* __restrict__ ln_b, int E_) {
  __shared__ float hs[16][DD];
  __shared__ float red[16][16][2];
  __shared__ float stats[16][2];

  const int tid = threadIdx.x, lane = tid & 31, wave = tid >> 5;
  const int rb = blockIdx.x;
  const int half = lane >> 4, lr = lane & 15;

  int arowi = rb * 16 + lr; if (arowi >= E_) arowi = E_ - 1;
  const _Float16* arow = hbuf + (size_t)arowi * DD + half * 16;

  v8f acc[4] = {};
  for (int kb = 0; kb < 16; ++kb) {
    v16h a;
    const v8h a0 = *(const v8h*)(arow + kb * 32);
    const v8h a1 = *(const v8h*)(arow + kb * 32 + 8);
    #pragma unroll
    for (int t = 0; t < 8; ++t) { a[t] = a0[t]; a[8 + t] = a1[t]; }
    #pragma unroll
    for (int i = 0; i < 4; ++i) {
      const int ct = wave * 4 + i;
      const _Float16* bp = W1T + (size_t)(ct * 16 + lr) * DD + kb * 32 + half * 16;
      v16h b = *(const v16h*)bp;
      acc[i] = wmma_f16(a, b, acc[i]);
    }
  }

  #pragma unroll
  for (int i = 0; i < 4; ++i) {
    const int col = (wave * 4 + i) * 16 + lr;
    const float bb = b1[col];
    #pragma unroll
    for (int v = 0; v < 8; ++v) {
      float val = acc[i][v] + bb;
      val = val / (1.f + __expf(-val));
      hs[v + half * 8][col] = val;
    }
  }
  __syncthreads();

  {
    const int row = tid >> 4, seg = tid & 15;
    float s = 0.f, s2 = 0.f;
    for (int j = 0; j < 32; ++j) {
      float v = hs[row][seg * 32 + j];
      s += v; s2 += v * v;
    }
    red[row][seg][0] = s; red[row][seg][1] = s2;
  }
  __syncthreads();
  if (tid < 16) {
    float s = 0.f, s2 = 0.f;
    for (int j = 0; j < 16; ++j) { s += red[tid][j][0]; s2 += red[tid][j][1]; }
    const float mu = s / (float)DD;
    const float var = s2 / (float)DD - mu * mu;
    stats[tid][0] = mu;
    stats[tid][1] = rsqrtf(var + 1e-5f);
  }
  __syncthreads();
  {
    const int row = tid >> 4, seg = tid & 15;
    const int er = rb * 16 + row;
    if (er < E_) {
      const float mu = stats[row][0], rs = stats[row][1];
      _Float16* orow = hbuf + (size_t)er * DD;
      for (int j = 0; j < 32; ++j) {
        const int c = seg * 32 + j;
        orow[c] = (_Float16)((hs[row][c] - mu) * rs * ln_g[c] + ln_b[c]);
      }
    }
  }
}

// ---------------------------------------------------------------------------
// Kernel 3: mlp = h2 @ W2 + b2; weight blocks stored (k,m)-swapped in LDS so
// final-matmul B-fragments are contiguous ds_load_b128; frob-norm folded in.
// ---------------------------------------------------------------------------
__global__ __launch_bounds__(256)
void gemm2_edge_kernel(const _Float16* __restrict__ h2,
                       const _Float16* __restrict__ W2T,
                       const float* __restrict__ b2,
                       const float* __restrict__ efa,
                       const float* __restrict__ efb,
                       const float* __restrict__ x,
                       const int* __restrict__ eidx,
                       float* __restrict__ out_a, float* __restrict__ out_b,
                       int E_) {
  __shared__ __align__(32) _Float16 wlds[16][1024]; // blocks stored transposed
  __shared__ float facs[16][64];

  const int tid = threadIdx.x, lane = tid & 31, wave = tid >> 5;
  const int rb = blockIdx.x;
  const int half = lane >> 4, lr = lane & 15;
  const int kb = half * 8;
  const int nt = (wave < 4) ? 9 : 8;

  int arowi = rb * 16 + lr; if (arowi >= E_) arowi = E_ - 1;
  const _Float16* arow = h2 + (size_t)arowi * DD + half * 16;

  v8f acc[9];
  #pragma unroll
  for (int i = 0; i < 9; ++i) acc[i] = (v8f){};

  for (int kk = 0; kk < 16; ++kk) {
    v16h a;
    const v8h a0 = *(const v8h*)(arow + kk * 32);
    const v8h a1 = *(const v8h*)(arow + kk * 32 + 8);
    #pragma unroll
    for (int t = 0; t < 8; ++t) { a[t] = a0[t]; a[8 + t] = a1[t]; }
    #pragma unroll
    for (int i = 0; i < 9; ++i) {
      if (i < nt) {
        const int ct = wave + 8 * i;
        const _Float16* bp = W2T + (size_t)(ct * 16 + lr) * DD + kk * 32 + half * 16;
        v16h b = *(const v16h*)bp;
        acc[i] = wmma_f16(a, b, acc[i]);
      }
    }
  }

  #pragma unroll
  for (int i = 0; i < 9; ++i) {
    if (i < nt) {
      const int cg = (wave + 8 * i) * 16 + lr;
      const float bb = b2[cg];
      // swap (k,m) within each 256-entry block: cg = b*256 + k*16 + m
      const int widx = (cg & 0x300) | ((cg & 15) << 4) | ((cg >> 4) & 15);
      #pragma unroll
      for (int v = 0; v < 8; ++v) {
        const float val = acc[i][v] + bb;
        const int row = v + half * 8;
        if (cg < 1024) wlds[row][widx] = (_Float16)val;
        else           facs[row][cg - 1024] = val;
      }
    }
  }
  __syncthreads();

  // A-fragment from global f32 row-major 16x16 (contiguous 8 floats/lane).
  auto fragA = [&](const float* m) -> v16h {
    const float4* p = (const float4*)(m + lr * 16 + kb);
    const float4 u0 = p[0], u1 = p[1];
    v16h f = {};
    f[0] = (_Float16)u0.x; f[1] = (_Float16)u0.y;
    f[2] = (_Float16)u0.z; f[3] = (_Float16)u0.w;
    f[4] = (_Float16)u1.x; f[5] = (_Float16)u1.y;
    f[6] = (_Float16)u1.z; f[7] = (_Float16)u1.w;
    return f;
  };
  // B-fragment from (k,m)-swapped f16 block: wb[m*16+k], elements 0..7 = wb[lr*16+kb..+7].
  auto fragBh = [&](const _Float16* wb) -> v16h {
    const v8h* p = (const v8h*)(wb + lr * 16 + kb);
    const v8h u0 = p[0];
    v16h f = {};
    #pragma unroll
    for (int t = 0; t < 8; ++t) { f[t] = u0[t]; }
    return f;
  };

  for (int ee = 0; ee < 2; ++ee) {
    const int eloc = wave * 2 + ee;
    const int e = rb * 16 + eloc;
    if (e >= E_) continue;

    float scale[4];
    #pragma unroll
    for (int b = 0; b < 4; ++b) {
      float s = 0.f;
      const v8h* wb = (const v8h*)&wlds[eloc][b * 256];
      const v8h u = wb[lane];
      #pragma unroll
      for (int j = 0; j < 8; ++j) { const float v = (float)u[j]; s += v * v; }
      #pragma unroll
      for (int m = 16; m >= 1; m >>= 1) s += __shfl_xor(s, m, 32);
      scale[b] = 1.f / (sqrtf(s) + 1e-6f);
    }

    const int ia = eidx[e], ib = eidx[E_ + e];
    const v8f z = {};

    v8f r1 = wmma_f16(fragA(efa + (size_t)e * 256), fragBh(&wlds[eloc][0]),   z);
    v8f r2 = wmma_f16(fragA(x  + (size_t)ia * 256), fragBh(&wlds[eloc][512]), z);
    const float fe = facs[eloc][0 + lr]  * scale[0];
    const float fn = facs[eloc][32 + lr] * scale[2];
    float* oa = out_a + (size_t)e * 256;
    #pragma unroll
    for (int v = 0; v < 8; ++v)
      oa[(v + half * 8) * 16 + lr] = 0.5f * (fe * r1[v] + fn * r2[v]);

    v8f r3 = wmma_f16(fragA(efb + (size_t)e * 256), fragBh(&wlds[eloc][256]), z);
    v8f r4 = wmma_f16(fragA(x  + (size_t)ib * 256), fragBh(&wlds[eloc][768]), z);
    const float fe2 = facs[eloc][16 + lr] * scale[1];
    const float fn2 = facs[eloc][48 + lr] * scale[3];
    float* ob = out_b + (size_t)e * 256;
    #pragma unroll
    for (int v = 0; v < 8; ++v)
      ob[(v + half * 8) * 16 + lr] = 0.5f * (fe2 * r3[v] + fn2 * r4[v]);
  }
}

// ---------------------------------------------------------------------------
extern "C" void kernel_launch(void* const* d_in, const int* in_sizes, int n_in,
                              void* d_out, int out_size, void* d_ws, size_t ws_size,
                              hipStream_t stream) {
  const float* x    = (const float*)d_in[0];
  const int*   eidx = (const int*)  d_in[1];
  const float* efa  = (const float*)d_in[2];
  const float* efb  = (const float*)d_in[3];
  const float* em   = (const float*)d_in[4];
  const float* W1   = (const float*)d_in[5];
  const float* b1   = (const float*)d_in[6];
  const float* lng  = (const float*)d_in[7];
  const float* lnb  = (const float*)d_in[8];
  const float* W2   = (const float*)d_in[9];
  const float* b2   = (const float*)d_in[10];
  float* out = (float*)d_out;
  const int E_ = in_sizes[1] / 2;

  char* ws = (char*)d_ws;
  _Float16* W1T  = (_Float16*)ws;
  _Float16* W2T  = (_Float16*)(ws + (size_t)DD * DD * 2);
  _Float16* hbuf = (_Float16*)(ws + (size_t)DD * DD * 2 + (size_t)NC2 * DD * 2);

  const int prep_total = DD * DD + NC2 * DD;
  prep_weights<<<(prep_total + 255) / 256, 256, 0, stream>>>(W1, W2, W1T, W2T);

  overlap_kernel<<<(E_ + 7) / 8, 256, 0, stream>>>(x, eidx, efa, efb, em, hbuf, E_);

  gemm1_ln_kernel<<<(E_ + 15) / 16, 256, 0, stream>>>(hbuf, W1T, b1, lng, lnb, E_);

  gemm2_edge_kernel<<<(E_ + 15) / 16, 256, 0, stream>>>(
      hbuf, W2T, b2, efa, efb, x, eidx, out, out + (size_t)E_ * 256, E_);
}
